// GCNPredictor_10488310137589
// MI455X (gfx1250) — matrix operand order repaired and checked by reading, not compiled, not run on hardware
//
#include <hip/hip_runtime.h>
#include <hip/hip_bf16.h>

// GCN on MI455X (gfx1250, wave32).
// Key sizing: feature matrices are n*64*4B = 25.6MB -> fit in 192MB L2, so the
// edge gather + float-atomic scatter phase is L2-resident; atomics are cheap.
// GEMMs use v_wmma_f32_16x16x32_f16 (f16 in, f32 accumulate).

typedef __attribute__((ext_vector_type(16))) _Float16 v16h;
typedef __attribute__((ext_vector_type(8)))  float    v8f;

// ---------------------------------------------------------------- degree
__global__ void k_deg_init(float* __restrict__ deg, int n) {
  int i = blockIdx.x * blockDim.x + threadIdx.x;
  if (i < n) deg[i] = 1.0f;                    // self-loop contributes 1
}

__global__ void k_deg_count(const int* __restrict__ dst, float* __restrict__ deg, int e) {
  int i = blockIdx.x * blockDim.x + threadIdx.x;
  if (i < e) atomicAdd(&deg[dst[i]], 1.0f);
}

__global__ void k_dinv(const float* __restrict__ deg, float* __restrict__ dinv, int n) {
  int i = blockIdx.x * blockDim.x + threadIdx.x;
  if (i < n) {
    float d = deg[i];
    dinv[i] = d > 0.0f ? rsqrtf(d) : 0.0f;
  }
}

// ------------------------------------------- WMMA GEMM: Y = dinv .* (X @ W)
// Also seeds ACC = Y (the self-loop term of the aggregation).
// One wave computes a 16-row x 64-col tile: 4 n-tiles x 2 k-steps = 8 WMMAs.
__global__ __launch_bounds__(128)
void k_gemm_scale(const float* __restrict__ X, const float* __restrict__ W,
                  const float* __restrict__ dinv, float* __restrict__ Y,
                  float* __restrict__ ACC, int n_rows)
{
  // W (64x64 f32, row-major) -> LDS transposed f16: WT[n*64 + k] = W[k*64 + n].
  // B fragment per lane = one column of W, two 8-half runs in WT, 16 apart (k..k+7, k+16..k+23).
  __shared__ __align__(32) _Float16 WT[64 * 64];
  for (int i = threadIdx.x; i < 64 * 64; i += 128) {
    int k = i >> 6, nn = i & 63;
    WT[nn * 64 + k] = (_Float16)W[i];
  }
  __syncthreads();

  const int lane = threadIdx.x & 31;
  const int wave = threadIdx.x >> 5;
  const int tile = blockIdx.x * 4 + wave;
  if (tile * 16 >= n_rows) return;              // wave-uniform: EXEC stays all-1
  const int m0   = tile << 4;
  const int half = lane >> 4;                   // 0: lanes 0-15, 1: lanes 16-31
  const int l16  = lane & 15;

  // A fragment (16x32 f16 per k-step): lane holds row m0+l16,
  //   elements 0-7  = K (ks*32 + half*8 + 0..7)
  //   elements 8-15 = K (ks*32 + half*8 + 16..23)
  int arow = m0 + l16; if (arow >= n_rows) arow = n_rows - 1;
  const float4* x4 = (const float4*)(X + (size_t)arow * 64);
  v16h a[2];
#pragma unroll
  for (int ks = 0; ks < 2; ++ks) {
    const int q = ks * 8 + half * 2;            // float4 index of first chunk
    float4 p0 = x4[q],     p1 = x4[q + 1];      // K = base .. base+7
    float4 p2 = x4[q + 4], p3 = x4[q + 5];      // K = base+16 .. base+23
    a[ks][0]  = (_Float16)p0.x; a[ks][1]  = (_Float16)p0.y;
    a[ks][2]  = (_Float16)p0.z; a[ks][3]  = (_Float16)p0.w;
    a[ks][4]  = (_Float16)p1.x; a[ks][5]  = (_Float16)p1.y;
    a[ks][6]  = (_Float16)p1.z; a[ks][7]  = (_Float16)p1.w;
    a[ks][8]  = (_Float16)p2.x; a[ks][9]  = (_Float16)p2.y;
    a[ks][10] = (_Float16)p2.z; a[ks][11] = (_Float16)p2.w;
    a[ks][12] = (_Float16)p3.x; a[ks][13] = (_Float16)p3.y;
    a[ks][14] = (_Float16)p3.z; a[ks][15] = (_Float16)p3.w;
  }

  v8f c[4] = {};
#pragma unroll
  for (int nt = 0; nt < 4; ++nt) {
#pragma unroll
    for (int ks = 0; ks < 2; ++ks) {
      // B fragment (32x16): lane holds column nt*16+l16, K = ks*32 + half*8 + 0..7 and ks*32 + half*8 + 16..23
      typedef __attribute__((ext_vector_type(8))) _Float16 v8h_;
      const _Float16* wp = &WT[(nt * 16 + l16) * 64 + ks * 32 + half * 8];
      const v8h_ b0 = *(const v8h_*)wp;
      const v8h_ b1 = *(const v8h_*)(wp + 16);
      const v16h b = __builtin_shufflevector(b0, b1, 0,1,2,3,4,5,6,7,8,9,10,11,12,13,14,15);
      c[nt] = __builtin_amdgcn_wmma_f32_16x16x32_f16(
          /*neg_a=*/false, a[ks], /*neg_b=*/false, b,
          /*c_mod=*/(short)0, c[nt], /*reuse_a=*/false, /*reuse_b=*/false);
    }
  }

  // D layout: VGPR r holds row (r + 8*half), col = l16 (+ nt*16).
#pragma unroll
  for (int r = 0; r < 8; ++r) {
    const int rr = m0 + r + half * 8;
    if (rr >= n_rows) continue;
    const float dv = dinv[rr];
    float* yrow = Y   + (size_t)rr * 64 + l16;
    float* crow = ACC + (size_t)rr * 64 + l16;
#pragma unroll
    for (int nt = 0; nt < 4; ++nt) {
      const float v = c[nt][r] * dv;
      yrow[nt * 16] = v;     // gather source
      crow[nt * 16] = v;     // accumulator seeded with self-loop term
    }
  }
}

// ------------------------------------- edge scatter: ACC[dst] += Y[src]
// 16 threads per edge, float4 per thread. Y/ACC are L2-resident (25.6MB each).
__global__ void k_scatter(const int* __restrict__ src, const int* __restrict__ dst,
                          const float* __restrict__ Y, float* __restrict__ ACC,
                          int n_edges)
{
  long long t = (long long)blockIdx.x * blockDim.x + threadIdx.x;
  int e = (int)(t >> 4);
  if (e >= n_edges) return;
  int c = ((int)t & 15) * 4;
  int s = src[e], d = dst[e];
  const float4 v = *(const float4*)&Y[(size_t)s * 64 + c];
  float* p = &ACC[(size_t)d * 64 + c];
  atomicAdd(p + 0, v.x);
  atomicAdd(p + 1, v.y);
  atomicAdd(p + 2, v.z);
  atomicAdd(p + 3, v.w);
}

// --------------------------- epilogue: H = relu(dinv[i] * ACC + bias)
__global__ void k_bias_relu(const float* __restrict__ ACC, const float* __restrict__ dinv,
                            const float* __restrict__ bias, float* __restrict__ H,
                            int n)
{
  long long t = (long long)blockIdx.x * blockDim.x + threadIdx.x;
  if (t >= (long long)n * 64) return;
  int i = (int)(t >> 6), f = (int)t & 63;
  float v = dinv[i] * ACC[t] + bias[f];
  H[t] = v > 0.0f ? v : 0.0f;
}

// ------------------------------------------------ global mean pool + FC
__global__ void k_pool_zero(float* __restrict__ gsum, float* __restrict__ gcnt, int G) {
  int i = blockIdx.x * blockDim.x + threadIdx.x;
  if (i < G * 64) gsum[i] = 0.0f;
  else if (i < G * 64 + G) gcnt[i - G * 64] = 0.0f;
}

__global__ void k_pool_acc(const float* __restrict__ H, const int* __restrict__ batch,
                           float* __restrict__ gsum, float* __restrict__ gcnt, int n)
{
  long long t = (long long)blockIdx.x * blockDim.x + threadIdx.x;
  if (t >= (long long)n * 64) return;
  int i = (int)(t >> 6), f = (int)t & 63;
  int g = batch[i];
  atomicAdd(&gsum[(size_t)g * 64 + f], H[t]);
  if (f == 0) atomicAdd(&gcnt[g], 1.0f);
}

// one wave32 per graph: out[g] = (sum_f gsum[g][f]*Wfc[f]) / max(cnt,1) + bfc
__global__ void k_fc(const float* __restrict__ gsum, const float* __restrict__ gcnt,
                     const float* __restrict__ Wfc, const float* __restrict__ bfc,
                     float* __restrict__ out)
{
  int g = blockIdx.x;
  int l = threadIdx.x;               // 0..31 (wave32)
  float v = gsum[(size_t)g * 64 + l]      * Wfc[l]
          + gsum[(size_t)g * 64 + 32 + l] * Wfc[32 + l];
#pragma unroll
  for (int off = 16; off > 0; off >>= 1) v += __shfl_down(v, off, 32);
  if (l == 0) out[g] = v / fmaxf(gcnt[g], 1.0f) + bfc[0];
}

// ----------------------------------------------------------------------
extern "C" void kernel_launch(void* const* d_in, const int* in_sizes, int n_in,
                              void* d_out, int out_size, void* d_ws, size_t ws_size,
                              hipStream_t stream)
{
  const float* x     = (const float*)d_in[0];
  const int*   ei    = (const int*)  d_in[1];   // [2, E] int32: row0=src, row1=dst
  const int*   batch = (const int*)  d_in[2];
  const float* W1    = (const float*)d_in[3];
  const float* b1    = (const float*)d_in[4];
  const float* W2    = (const float*)d_in[5];
  const float* b2    = (const float*)d_in[6];
  const float* Wfc   = (const float*)d_in[7];
  const float* bfc   = (const float*)d_in[8];
  float* out = (float*)d_out;

  const int n = in_sizes[0] / 64;
  const int E = in_sizes[1] / 2;
  const int G = out_size;
  const int* src = ei;
  const int* dstv = ei + E;

  // Workspace carve-out (floats), 64-float aligned. Total ~= 194*n*4 B ~ 77.7MB.
  float* ws = (float*)d_ws;
  size_t off = 0;
  auto alloc = [&](size_t cnt) {
    float* p = ws + off;
    off += (cnt + 63) & ~(size_t)63;
    return p;
  };
  float* deg  = alloc((size_t)n);
  float* dinv = alloc((size_t)n);
  float* bufY = alloc((size_t)n * 64);   // scaled GEMM output (gather source)
  float* bufA = alloc((size_t)n * 64);   // scatter accumulator
  float* bufH = alloc((size_t)n * 64);   // layer-1 activations
  float* gsum = alloc((size_t)G * 64);
  float* gcnt = alloc((size_t)G);
  (void)ws_size; (void)n_in;

  const int T = 256;
  const int nBlk  = (n + T - 1) / T;
  const int eBlk  = (E + T - 1) / T;
  const long long nf = (long long)n * 64;
  const int nfBlk = (int)((nf + T - 1) / T);
  const long long et = (long long)E * 16;
  const int etBlk = (int)((et + T - 1) / T);
  const int tiles = (n + 15) / 16;
  const int gemmBlk = (tiles + 3) / 4;          // 4 waves / block

  // degrees (in-degree incl. self loop) and dinv = deg^-1/2
  k_deg_init <<<nBlk, T, 0, stream>>>(deg, n);
  k_deg_count<<<eBlk, T, 0, stream>>>(dstv, deg, E);
  k_dinv     <<<nBlk, T, 0, stream>>>(deg, dinv, n);

  // ---- layer 1: h1 = relu(dinv*(scatter(dinv*(x@W1))) + b1)
  k_gemm_scale<<<gemmBlk, 128, 0, stream>>>(x, W1, dinv, bufY, bufA, n);
  k_scatter   <<<etBlk, T, 0, stream>>>(src, dstv, bufY, bufA, E);
  k_bias_relu <<<nfBlk, T, 0, stream>>>(bufA, dinv, b1, bufH, n);

  // ---- layer 2 (reuse Y/A; h2 lands in bufY once scatter is done with it)
  k_gemm_scale<<<gemmBlk, 128, 0, stream>>>(bufH, W2, dinv, bufY, bufA, n);
  k_scatter   <<<etBlk, T, 0, stream>>>(src, dstv, bufY, bufA, E);
  k_bias_relu <<<nfBlk, T, 0, stream>>>(bufA, dinv, b2, bufY, n);   // h2 = bufY

  // ---- mean pool + FC
  k_pool_zero<<<(G * 65 + T - 1) / T, T, 0, stream>>>(gsum, gcnt, G);
  k_pool_acc <<<nfBlk, T, 0, stream>>>(bufY, batch, gsum, gcnt, n);
  k_fc       <<<G, 32, 0, stream>>>(gsum, gcnt, Wfc, bfc, out);
}
